// ManifoldAutomata_76965813944494
// MI455X (gfx1250) — matrix operand order repaired and checked by reading, not compiled, not run on hardware
//
#include <hip/hip_runtime.h>
#include <hip/hip_bf16.h>

typedef __attribute__((ext_vector_type(16))) _Float16 v16h;
typedef __attribute__((ext_vector_type(8)))  float    v8f;

#define WMMA_F16(A, B, C) \
  __builtin_amdgcn_wmma_f32_16x16x32_f16(false, (A), false, (B), (short)0, (C), false, false)

// A-fragment K index for 16-bit A (16x32): VGPR v (pair 2v,2v+1), lane-half h.
__device__ __forceinline__ int a_kbase(int v, int half) {
  return ((v & 4) << 2) + ((v & 3) << 1) + (half << 3);
}

// ---------------------------------------------------------------------------
// prep: f32 x -> f16 copy + per-row sum of squares
// ---------------------------------------------------------------------------
__global__ void prep_kernel(const float* __restrict__ x, _Float16* __restrict__ xh,
                            float* __restrict__ sq, int rows) {
  int r = blockIdx.x * blockDim.x + threadIdx.x;
  if (r >= rows) return;
  const float4* p = (const float4*)(x + (size_t)r * 64);
  _Float16* o = xh + (size_t)r * 64;
  float ss = 0.f;
#pragma unroll
  for (int q = 0; q < 16; ++q) {
    float4 v = p[q];
    ss += v.x * v.x + v.y * v.y + v.z * v.z + v.w * v.w;
    o[4 * q + 0] = (_Float16)v.x;
    o[4 * q + 1] = (_Float16)v.y;
    o[4 * q + 2] = (_Float16)v.z;
    o[4 * q + 3] = (_Float16)v.w;
  }
  sq[r] = ss;
}

// ---------------------------------------------------------------------------
// Fused N x N pass.  Per (b, 64-row i-block), loop j in chunks of 32:
//   x_j chunk staged into LDS via global_load_async_to_lds_b128 (ASYNCcnt)
//   G = X_i X_j^T via f16 WMMA (K=64)
//   CORR: heb_new = .95*heb + .01 * G * rn_i * rn_j  -> heb_out
//   AGG : EA = exp(-sqrt(relu(sq_i+sq_j-2G))) + .1*heb_new
//         agg_i += EA @ X_j  via f16 WMMA (K=32)          -> agg / (N+eps)
// ---------------------------------------------------------------------------
template <bool CORR, bool AGG>
__global__ void manifold_pass(const _Float16* __restrict__ xh, const float* __restrict__ sq,
                              const float* __restrict__ rn, const float* heb_in,
                              float* heb_out, float* agg, int N, float invNeps) {
  const int b = blockIdx.y;
  const int i0 = blockIdx.x * 64;
  const int tid = threadIdx.x;
  const int lane = tid & 31;
  const int wv = tid >> 5;     // 4 waves / block, 16 i-rows each
  const int half = lane >> 4;
  const int lm = lane & 15;

  __shared__ alignas(16) _Float16 lxr[32][72];      // x_j  [row][col]  (gram B frags)
  __shared__ alignas(16) _Float16 lxt[64][40];      // x_j^T [col][row] (agg  B frags)
  __shared__ float lsq[32];
  __shared__ float lrn[32];
  __shared__ alignas(16) _Float16 lea[4][16][40];   // per-wave EA staging [m][k]

  const size_t bN = (size_t)b * N;

  float sqi[8], rni[8];
#pragma unroll
  for (int v = 0; v < 8; ++v) {
    int ir = i0 + wv * 16 + v + 8 * half;
    sqi[v] = sq[bN + ir];
    if (CORR) rni[v] = rn[bN + ir];
  }

  // persistent A fragments of X_i (16 x 64 f16, two K=32 frags)
  v16h axi[2];
  {
    const _Float16* xr = xh + (bN + i0 + wv * 16 + lm) * 64;
#pragma unroll
    for (int f = 0; f < 2; ++f)
#pragma unroll
      for (int v = 0; v < 8; ++v) {
        int k = a_kbase(v, half) + 32 * f;
        axi[f][2 * v] = xr[k];
        axi[f][2 * v + 1] = xr[k + 1];
      }
  }

  v8f acc[4];
  if (AGG) {
#pragma unroll
    for (int s = 0; s < 4; ++s)
#pragma unroll
      for (int e = 0; e < 8; ++e) acc[s][e] = 0.f;
  }

  const unsigned lxr_base = (unsigned)(uintptr_t)&lxr[0][0];

  for (int j0 = 0; j0 < N; j0 += 32) {
    __syncthreads();  // previous chunk fully consumed

    // ---- async stage of X_j chunk (32 rows x 64 f16 = 4KB), 16B per lane ----
    {
#pragma unroll
      for (int it = 0; it < 2; ++it) {
        int g = tid + 128 * it;          // 256 segments of 8 f16
        int r = g >> 3;
        int c0 = (g & 7) << 3;
        unsigned loff = lxr_base + (unsigned)(r * 72 + c0) * 2u;
        unsigned long long ga =
            (unsigned long long)(uintptr_t)(xh + (bN + j0 + r) * 64 + c0);
        asm volatile("global_load_async_to_lds_b128 %0, %1, off"
                     :: "v"(loff), "v"(ga) : "memory");
      }
      if (tid < 32) {
        lsq[tid] = sq[bN + j0 + tid];
        if (CORR) lrn[tid] = rn[bN + j0 + tid];
      }
      asm volatile("s_wait_asynccnt 0x0" ::: "memory");
    }
    __syncthreads();

    if (AGG) {  // build transposed copy for agg B-fragments
#pragma unroll
      for (int it = 0; it < 2; ++it) {
        int g = tid + 128 * it;
        int r = g >> 3;
        int c0 = (g & 7) << 3;
#pragma unroll
        for (int c = 0; c < 8; ++c) lxt[c0 + c][r] = lxr[r][c0 + c];
      }
      __syncthreads();
    }

    // prefetch next chunk's heb tile (64 rows x 128B) while we compute
    {
      int jn = j0 + 32;
      if (jn < N) {
        int r = tid >> 1, sgm = tid & 1;
        __builtin_prefetch(heb_in + (bN + i0 + r) * (size_t)N + jn + sgm * 16, 0, 1);
      }
    }

    // gram tiles: two 16x16 outputs (j and j+16)
    v8f g[2];
#pragma unroll
    for (int t = 0; t < 2; ++t) {
#pragma unroll
      for (int e = 0; e < 8; ++e) g[t][e] = 0.f;
#pragma unroll
      for (int f = 0; f < 2; ++f) {
        v16h bg;  // B[k][n] = X_j[16t+n][k], K = D-range
#pragma unroll
        for (int v = 0; v < 8; ++v) {
          int k = a_kbase(v, half) + 32 * f;
          bg[2 * v] = lxr[16 * t + lm][k];
          bg[2 * v + 1] = lxr[16 * t + lm][k + 1];
        }
        g[t] = WMMA_F16(axi[f], bg, g[t]);
      }
    }

    // elementwise: heb update / effective adjacency
#pragma unroll
    for (int t = 0; t < 2; ++t) {
#pragma unroll
      for (int v = 0; v < 8; ++v) {
        int m = v + 8 * half;                 // C-layout row
        int ir = i0 + wv * 16 + m;
        int jc = j0 + 16 * t + lm;
        size_t hidx = (bN + ir) * (size_t)N + jc;
        float go = g[t][v];
        float hnew = heb_in[hidx];
        if (CORR) {
          hnew = 0.95f * hnew + 0.01f * (go * rni[v] * lrn[16 * t + lm]);
          heb_out[hidx] = hnew;
        }
        if (AGG) {
          float d2 = fmaxf(sqi[v] + lsq[16 * t + lm] - 2.0f * go, 0.f);
          float av = __expf(-__builtin_amdgcn_sqrtf(d2));
          lea[wv][m][16 * t + lm] = (_Float16)(av + 0.1f * hnew);
        }
      }
    }
    __syncthreads();

    if (AGG) {  // agg += EA(16x32) @ X_j(32x64)
      v16h ae;
#pragma unroll
      for (int v = 0; v < 8; ++v) {
        int k = a_kbase(v, half);
        ae[2 * v] = lea[wv][lm][k];
        ae[2 * v + 1] = lea[wv][lm][k + 1];
      }
#pragma unroll
      for (int s = 0; s < 4; ++s) {
        v16h bA;  // B[k][n] = X_j[k][16s+n], K = j-range
#pragma unroll
        for (int v = 0; v < 8; ++v) {
          int k = a_kbase(v, half);
          bA[2 * v] = lxt[s * 16 + lm][k];
          bA[2 * v + 1] = lxt[s * 16 + lm][k + 1];
        }
        acc[s] = WMMA_F16(ae, bA, acc[s]);
      }
    }
  }

  if (AGG) {
#pragma unroll
    for (int s = 0; s < 4; ++s)
#pragma unroll
      for (int v = 0; v < 8; ++v) {
        int ir = i0 + wv * 16 + v + 8 * half;
        agg[(bN + ir) * 64 + s * 16 + lm] = acc[s][v] * invNeps;
      }
  }
}

// ---------------------------------------------------------------------------
// Per-row MLP: [x,agg]@W1+b1 -> LN -> silu -> @W2+b2 -> tanh -> x+=delta.
// Also emits f16 x, row sumsq and 1/max(||x||,1e-12) for the next pass.
// 32 rows / 256-thread block; weights + activations in dynamic LDS (~138 KB).
// ---------------------------------------------------------------------------
__global__ void mlp_kernel(const float* __restrict__ x, const float* __restrict__ agg,
                           const float* __restrict__ W1, const float* __restrict__ b1,
                           const float* __restrict__ gam, const float* __restrict__ bet,
                           const float* __restrict__ W2, const float* __restrict__ b2,
                           float* __restrict__ xout, _Float16* __restrict__ xh,
                           float* __restrict__ sqo, float* __restrict__ rno) {
  extern __shared__ char smemraw[];
  float* sW1 = (float*)smemraw;        // 128*128
  float* sW2 = sW1 + 128 * 128;        // 128*64
  float* sb1 = sW2 + 128 * 64;
  float* sg = sb1 + 128;
  float* sbt = sg + 128;
  float* sb2 = sbt + 128;
  float* scomb = sb2 + 64;             // 32*129
  float* sact = scomb + 32 * 129;      // 32*129
  float* sxn = sact + 32 * 129;        // 32*65
  float* smu = sxn + 32 * 65;          // 32
  float* srs = smu + 32;               // 32

  const int t = threadIdx.x;
  const long long row0 = (long long)blockIdx.x * 32;

  for (int i = t; i < 128 * 128; i += 256) sW1[i] = W1[i];
  for (int i = t; i < 128 * 64; i += 256) sW2[i] = W2[i];
  if (t < 128) { sb1[t] = b1[t]; sg[t] = gam[t]; sbt[t] = bet[t]; }
  if (t < 64) sb2[t] = b2[t];
  for (int idx = t; idx < 32 * 128; idx += 256) {
    int r = idx >> 7, c = idx & 127;
    scomb[r * 129 + c] = (c < 64) ? x[(row0 + r) * 64 + c] : agg[(row0 + r) * 64 + (c - 64)];
  }
  __syncthreads();

  for (int idx = t; idx < 32 * 128; idx += 256) {
    int r = idx >> 7, o = idx & 127;
    float a = sb1[o];
    for (int i = 0; i < 128; ++i) a = fmaf(scomb[r * 129 + i], sW1[i * 128 + o], a);
    sact[r * 129 + o] = a;
  }
  __syncthreads();

  if (t < 32) {
    float mu = 0.f;
    for (int o = 0; o < 128; ++o) mu += sact[t * 129 + o];
    mu *= (1.0f / 128.0f);
    float var = 0.f;
    for (int o = 0; o < 128; ++o) { float d = sact[t * 129 + o] - mu; var += d * d; }
    var *= (1.0f / 128.0f);
    smu[t] = mu;
    srs[t] = rsqrtf(var + 1e-5f);
  }
  __syncthreads();

  for (int idx = t; idx < 32 * 128; idx += 256) {
    int r = idx >> 7, o = idx & 127;
    float v = (sact[r * 129 + o] - smu[r]) * srs[r] * sg[o] + sbt[o];
    sact[r * 129 + o] = v / (1.0f + __expf(-v));   // silu(LN(h))
  }
  __syncthreads();

  for (int idx = t; idx < 32 * 64; idx += 256) {
    int r = idx >> 6, d = idx & 63;
    float a = sb2[d];
    for (int o = 0; o < 128; ++o) a = fmaf(sact[r * 129 + o], sW2[o * 64 + d], a);
    float xv = x[(row0 + r) * 64 + d] + tanhf(a);
    sxn[r * 65 + d] = xv;
    long long gi = (row0 + r) * 64 + d;
    xout[gi] = xv;
    xh[gi] = (_Float16)xv;
  }
  __syncthreads();

  if (t < 32) {
    float ss = 0.f;
    for (int d = 0; d < 64; ++d) { float v = sxn[t * 65 + d]; ss += v * v; }
    float nrm = fmaxf(sqrtf(ss), 1e-12f);
    sqo[row0 + t] = ss;
    rno[row0 + t] = 1.0f / nrm;
  }
}

// ---------------------------------------------------------------------------
extern "C" void kernel_launch(void* const* d_in, const int* in_sizes, int n_in,
                              void* d_out, int out_size, void* d_ws, size_t ws_size,
                              hipStream_t stream) {
  const float* x0 = (const float*)d_in[0];
  const float* heb0 = (const float*)d_in[1];
  const float* W1 = (const float*)d_in[2];
  const float* b1 = (const float*)d_in[3];
  const float* gam = (const float*)d_in[4];
  const float* bet = (const float*)d_in[5];
  const float* W2 = (const float*)d_in[6];
  const float* b2 = (const float*)d_in[7];
  (void)n_in; (void)out_size; (void)ws_size;

  const int D = 64;
  const long long BN = (long long)in_sizes[0] / D;        // B*N rows
  const int N = (int)((long long)in_sizes[1] / BN);       // (B*N*N)/(B*N)
  const int B = (int)(BN / N);
  const float invNeps = 1.0f / ((float)N + 1e-5f);

  char* w = (char*)d_ws;
  _Float16* xhbuf = (_Float16*)w; w += (size_t)BN * D * sizeof(_Float16);
  float* sqb = (float*)w;  w += (size_t)BN * sizeof(float);
  float* rnb = (float*)w;  w += (size_t)BN * sizeof(float);
  float* aggb = (float*)w; w += (size_t)BN * D * sizeof(float);
  float* x1 = (float*)w;   w += (size_t)BN * D * sizeof(float);

  float* out_x = (float*)d_out;
  float* out_heb = out_x + (size_t)BN * D;

  dim3 pgrid((unsigned)(N / 64), (unsigned)B);
  size_t mlp_smem =
      (size_t)(128 * 128 + 128 * 64 + 3 * 128 + 64 + 2 * 32 * 129 + 32 * 65 + 64) * sizeof(float);
  unsigned mlp_blocks = (unsigned)(BN / 32);

  // step 1: EA1 = adj(x0)+.1*heb0 -> agg1
  prep_kernel<<<dim3((unsigned)((BN + 255) / 256)), 256, 0, stream>>>(x0, xhbuf, sqb, (int)BN);
  manifold_pass<false, true><<<pgrid, 128, 0, stream>>>(xhbuf, sqb, nullptr, heb0, nullptr,
                                                        aggb, N, invNeps);
  mlp_kernel<<<mlp_blocks, 256, mlp_smem, stream>>>(x0, aggb, W1, b1, gam, bet, W2, b2,
                                                    x1, xhbuf, sqb, rnb);
  // step 2 head fused with step-1 heb update:
  //   heb1 = .95*heb0+.01*corr(x1) -> out_heb; EA2 = adj(x1)+.1*heb1 -> agg2
  manifold_pass<true, true><<<pgrid, 128, 0, stream>>>(xhbuf, sqb, rnb, heb0, out_heb,
                                                       aggb, N, invNeps);
  mlp_kernel<<<mlp_blocks, 256, mlp_smem, stream>>>(x1, aggb, W1, b1, gam, bet, W2, b2,
                                                    out_x, xhbuf, sqb, rnb);
  // final heb update: heb2 = .95*heb1 + .01*corr(x2), in-place in d_out
  manifold_pass<true, false><<<pgrid, 128, 0, stream>>>(xhbuf, sqb, rnb, out_heb, out_heb,
                                                        nullptr, N, invNeps);
}
